// DynamicLlamaAttention_26044681682964
// MI455X (gfx1250) — matrix operand order repaired and checked by reading, not compiled, not run on hardware
//
#include <hip/hip_runtime.h>
#include <hip/hip_bf16.h>
#include <math.h>

#define SEQ      1024
#define BATCH    2
#define DMODEL   3072
#define KDIM     1024
#define HEADS    32
#define QHD      96
#define THR      0.95f
#define REPS_EPS 1e-8f

// LDS B-panel: 64 rows x 64 halves, padded to 72-half row stride (bank-conflict free)
#define BK        64
#define BROWS     64
#define BSTRIDE   72
#define BPANEL_H  (BROWS * BSTRIDE)      // halves per buffer (9216 B)

typedef __attribute__((ext_vector_type(16))) _Float16 v16h;
typedef __attribute__((ext_vector_type(8)))  _Float16 v8h;
typedef __attribute__((ext_vector_type(8)))  float    v8f;
typedef __attribute__((ext_vector_type(4)))  unsigned int v4u;
typedef __attribute__((ext_vector_type(8)))  unsigned int v8u;

// ---------------------------------------------------------------------------
// WMMA fragment loads per CDNA5 ISA 7.12.2 (wave32, 16x16x32 f16, f32 accum)
// ---------------------------------------------------------------------------
// A (16x32, MxK) from global: lanes 0-15 -> M=lane, halves {K0..7, K16..23};
//                             lanes 16-31 -> M=lane-16, halves {K8..15, K24..31}
__device__ __forceinline__ v16h load_a_frag(const _Float16* __restrict__ A, int lda,
                                            int m0, int k0, int lane) {
  int m  = m0 + (lane & 15);
  int kb = k0 + ((lane >> 4) << 3);
  const _Float16* p = A + (size_t)m * lda + kb;
  v8h lo = *(const v8h*)p;          // K = kb .. kb+7
  v8h hi = *(const v8h*)(p + 16);   // K = kb+16 .. kb+23
  v16h r;
#pragma unroll
  for (int j = 0; j < 8; ++j) { r[j] = lo[j]; r[j + 8] = hi[j]; }
  return r;
}

// B (32x16, KxN) from the padded LDS panel (Bs[n][k], K contiguous, stride 72):
// lanes 0-15 -> N=lane, K={kb..kb+7, kb+16..kb+23}; lanes 16-31 -> N=lane-16, K={kb+8..kb+15, kb+24..kb+31}
__device__ __forceinline__ v16h load_b_frag_lds(const _Float16* Bs,
                                                int n0, int kb, int lane) {
  int n = n0 + (lane & 15);
  int k = kb + ((lane >> 4) << 3);
  const _Float16* p = Bs + n * BSTRIDE + k;   // 16B-aligned (144B row stride)
  v8h lo = *(const v8h*)p;
  v8h hi = *(const v8h*)(p + 16);
  v16h r;
#pragma unroll
  for (int j = 0; j < 8; ++j) { r[j] = lo[j]; r[j + 8] = hi[j]; }
  return r;
}

// C/D (16x16 f32): lane holds N = lane&15; VGPR r holds M = r + 8*(lane>>4)
__device__ __forceinline__ void store_c_tile(float* __restrict__ Cf,
                                             _Float16* __restrict__ Ch, int ldc,
                                             int m0, int n0, int lane, v8f c,
                                             const float* __restrict__ bias, float scale) {
  int n  = n0 + (lane & 15);
  int mb = m0 + ((lane >> 4) << 3);
  float bv = bias ? bias[n] : 0.0f;
#pragma unroll
  for (int r = 0; r < 8; ++r) {
    float v = c[r] * scale + bv;
    size_t off = (size_t)(mb + r) * ldc + n;
    if (Cf) Cf[off] = v;
    if (Ch) Ch[off] = (_Float16)v;
  }
}

// ---------------------------------------------------------------------------
// Tensor Data Mover: issue a 2-D tile load (ISA 08, D# groups 0/1) into LDS.
// 2-byte elements, tile = tile0 halves x tile1 rows, zero-fill past tensor
// dims, LDS padding of 4 dwords after every 32 dwords (-> 72-half row stride).
// ---------------------------------------------------------------------------
__device__ __forceinline__ void tdm_load_b_panel(unsigned lds_off,
                                                 const _Float16* gptr,
                                                 unsigned tdim0, unsigned tdim1,
                                                 unsigned stride0) {
  unsigned long long ga = (unsigned long long)(uintptr_t)gptr;
  v4u g0;
  g0[0] = 1u;                                          // count=1, user descriptor
  g0[1] = lds_off;                                     // LDS byte address
  g0[2] = (unsigned)(ga & 0xffffffffull);              // global_addr[31:0]
  g0[3] = (unsigned)((ga >> 32) & 0x01ffffffull) | (2u << 30);  // addr[56:32], type=2
  v8u g1;
  g1[0] = (1u << 16)        // data_size = 2 bytes
        | (1u << 20)        // pad_enable
        | (4u << 22)        // pad_interval: 32 dwords
        | (3u << 25);       // pad_amount: 4 dwords
  g1[1] = (tdim0 & 0xffffu) << 16;                               // tensor_dim0 lo
  g1[2] = ((tdim0 >> 16) & 0xffffu) | ((tdim1 & 0xffffu) << 16); // dim0 hi | dim1 lo
  g1[3] = ((tdim1 >> 16) & 0xffffu) | ((unsigned)BK << 16);      // dim1 hi | tile_dim0
  g1[4] = (unsigned)BROWS;                                       // tile_dim1 (tile_dim2=0)
  g1[5] = stride0;                                               // tensor_dim0_stride lo
  g1[6] = 0u;                                                    // stride0 hi | stride1 lo
  g1[7] = 0u;
  asm volatile("tensor_load_to_lds %0, %1" :: "s"(g0), "s"(g1) : "memory");
}

// ---------------------------------------------------------------------------
// Batched WMMA GEMM with TDM-staged B panels:
//   C[z] = A[z] (MxK row-major) * Bt[z]^T   (Bt stored NxK row-major)
// 4 waves/block in a 2x2 grid; wave tile 64x32 (4x2 WMMA tiles); block 128x64.
// B panel (64 rows x 64 halves) double-buffered in LDS via tensor_load_to_lds.
// M % 128 == 0, N % 64 == 0, any K (TDM zero-fills the K tail).
// ---------------------------------------------------------------------------
__global__ __launch_bounds__(128) void gemm_wmma_tdm(
    const _Float16* __restrict__ A,  size_t aZ, int lda,
    const _Float16* __restrict__ Bt, size_t bZ, int ldb,
    const float* __restrict__ bias, float scale,
    float* __restrict__ Cf, size_t cZ,
    _Float16* __restrict__ Ch, size_t chZ, int ldc,
    int K) {
  __shared__ _Float16 Bs[2][BPANEL_H];

  const int lane = threadIdx.x & 31;
  const int w    = threadIdx.x >> 5;
  const int z    = blockIdx.z;
  const _Float16* Az = A  + aZ * z;
  const _Float16* Bz = Bt + bZ * z + (size_t)(blockIdx.x * 64) * ldb;
  const int m0 = blockIdx.y * 128 + (w >> 1) * 64;   // wave's first output row
  const int nw = (w & 1) * 32;                        // wave's n offset in block

  const unsigned ldsBase = (unsigned)(uintptr_t)(&Bs[0][0]);
  const int nCh = (K + BK - 1) / BK;

  if (w == 0)
    tdm_load_b_panel(ldsBase, Bz, (unsigned)K, 64u, (unsigned)ldb);

  v8f acc[4][2];
#pragma unroll
  for (int i = 0; i < 4; ++i)
#pragma unroll
    for (int j = 0; j < 2; ++j) acc[i][j] = (v8f){};

  for (int c = 0; c < nCh; ++c) {
    const int buf = c & 1;
    const int k0  = c * BK;
    if (w == 0) {
      if (c + 1 < nCh) {
        tdm_load_b_panel(ldsBase + (unsigned)((buf ^ 1) * BPANEL_H * 2),
                         Bz + (size_t)(c + 1) * BK,
                         (unsigned)(K - (c + 1) * BK), 64u, (unsigned)ldb);
        __builtin_amdgcn_s_wait_tensorcnt(1);   // panel c done (in-order), c+1 in flight
      } else {
        __builtin_amdgcn_s_wait_tensorcnt(0);
      }
    }
    __syncthreads();                            // publish panel c to all waves

#pragma unroll
    for (int kk = 0; kk < 2; ++kk) {
      const int kG = k0 + kk * 32;              // A reads past K hit zero-filled B
      v16h a0 = load_a_frag(Az, lda, m0,      kG, lane);
      v16h a1 = load_a_frag(Az, lda, m0 + 16, kG, lane);
      v16h a2 = load_a_frag(Az, lda, m0 + 32, kG, lane);
      v16h a3 = load_a_frag(Az, lda, m0 + 48, kG, lane);
      v16h b0 = load_b_frag_lds(Bs[buf], nw,      kk * 32, lane);
      v16h b1 = load_b_frag_lds(Bs[buf], nw + 16, kk * 32, lane);
      acc[0][0] = __builtin_amdgcn_wmma_f32_16x16x32_f16(false, a0, false, b0, (short)0, acc[0][0], false, false);
      acc[0][1] = __builtin_amdgcn_wmma_f32_16x16x32_f16(false, a0, false, b1, (short)0, acc[0][1], false, false);
      acc[1][0] = __builtin_amdgcn_wmma_f32_16x16x32_f16(false, a1, false, b0, (short)0, acc[1][0], false, false);
      acc[1][1] = __builtin_amdgcn_wmma_f32_16x16x32_f16(false, a1, false, b1, (short)0, acc[1][1], false, false);
      acc[2][0] = __builtin_amdgcn_wmma_f32_16x16x32_f16(false, a2, false, b0, (short)0, acc[2][0], false, false);
      acc[2][1] = __builtin_amdgcn_wmma_f32_16x16x32_f16(false, a2, false, b1, (short)0, acc[2][1], false, false);
      acc[3][0] = __builtin_amdgcn_wmma_f32_16x16x32_f16(false, a3, false, b0, (short)0, acc[3][0], false, false);
      acc[3][1] = __builtin_amdgcn_wmma_f32_16x16x32_f16(false, a3, false, b1, (short)0, acc[3][1], false, false);
    }
    __syncthreads();                            // reads of buf done before its reuse
  }

  float*    cf = Cf ? Cf + cZ  * z : nullptr;
  _Float16* ch = Ch ? Ch + chZ * z : nullptr;
  const int n0 = blockIdx.x * 64 + nw;
#pragma unroll
  for (int i = 0; i < 4; ++i)
#pragma unroll
    for (int j = 0; j < 2; ++j)
      store_c_tile(cf, ch, ldc, m0 + i * 16, n0 + j * 16, lane, acc[i][j], bias, scale);
}

// ---------------------------------------------------------------------------
// Elementwise conversions
// ---------------------------------------------------------------------------
__global__ void cvt_f32_to_f16(const float* __restrict__ x, _Float16* __restrict__ y, size_t n) {
  size_t i = (size_t)blockIdx.x * blockDim.x + threadIdx.x;
  size_t stride = (size_t)gridDim.x * blockDim.x;
  for (; i < n; i += stride) y[i] = (_Float16)x[i];
}

// W (K x N, row-major) -> Wt (N x K, row-major) in f16
__global__ void cvt_transpose_f16(const float* __restrict__ W, _Float16* __restrict__ Wt,
                                  int K, int N) {
  size_t i = (size_t)blockIdx.x * blockDim.x + threadIdx.x;
  size_t total = (size_t)K * N;
  size_t stride = (size_t)gridDim.x * blockDim.x;
  for (; i < total; i += stride) {
    int k = (int)(i / N), n = (int)(i % N);
    Wt[(size_t)n * K + k] = (_Float16)W[i];
  }
}

// ---------------------------------------------------------------------------
// Row L2-normalize (for cosine similarity): Y = X / max(||X||, eps), f16 out
// ---------------------------------------------------------------------------
__global__ __launch_bounds__(256) void normalize_rows(const float* __restrict__ X,
                                                      _Float16* __restrict__ Y) {
  const int row = blockIdx.x;                  // 0 .. BATCH*SEQ-1
  const float* x = X + (size_t)row * KDIM;
  __shared__ float red[256];
  float s = 0.0f;
  for (int c = threadIdx.x; c < KDIM; c += 256) { float v = x[c]; s += v * v; }
  red[threadIdx.x] = s; __syncthreads();
  for (int st = 128; st > 0; st >>= 1) {
    if (threadIdx.x < st) red[threadIdx.x] += red[threadIdx.x + st];
    __syncthreads();
  }
  float inv = 1.0f / fmaxf(sqrtf(red[0]), REPS_EPS);
  for (int c = threadIdx.x; c < KDIM; c += 256)
    Y[(size_t)row * KDIM + c] = (_Float16)(x[c] * inv);
}

// ---------------------------------------------------------------------------
// Greedy sequential token-merge scan (matches the lax.scan semantics).
// One block of SEQ threads per (tensor,batch) slab; sim is SxS f32.
// ---------------------------------------------------------------------------
__global__ __launch_bounds__(1024) void greedy_merge_scan(const float* __restrict__ simBase,
                                                          int* __restrict__ repBase) {
  const int slab = blockIdx.x;                       // 0..3 : (tensor*2 + b)
  const float* sim = simBase + (size_t)slab * SEQ * SEQ;
  int* rep = repBase + (size_t)slab * SEQ;
  __shared__ unsigned char used[SEQ];
  __shared__ int repS[SEQ];
  const int j = threadIdx.x;
  used[j] = 0; repS[j] = j;
  __syncthreads();
  for (int i = 0; i < SEQ; ++i) {
    // writers this iteration have index j>i, so used[i]/used[j] reads are race-free
    if (!used[i] && j > i && !used[j] && sim[(size_t)i * SEQ + j] > THR) {
      used[j] = 1; repS[j] = i;
    }
    __syncthreads();
  }
  rep[j] = repS[j];
}

// new_k[b][t][:] = (f16) k[b][rep[t]][:]   (row-major: directly usable as Bt for scores)
__global__ void gather_rows_f16(const float* __restrict__ X, const int* __restrict__ rep,
                                _Float16* __restrict__ Y) {
  size_t i = (size_t)blockIdx.x * blockDim.x + threadIdx.x;
  const size_t total = (size_t)BATCH * SEQ * KDIM;
  if (i >= total) return;
  int c = (int)(i % KDIM);
  size_t bt = i / KDIM;
  int t = (int)(bt % SEQ), b = (int)(bt / SEQ);
  int r = rep[b * SEQ + t];
  Y[i] = (_Float16)X[((size_t)b * SEQ + r) * KDIM + c];
}

// new_vT[b][n][t] = (f16) v[b][rep[t]][n]  (N-major: directly usable as Bt for PV GEMM)
__global__ void gather_rows_T_f16(const float* __restrict__ X, const int* __restrict__ rep,
                                  _Float16* __restrict__ Y) {
  size_t i = (size_t)blockIdx.x * blockDim.x + threadIdx.x;
  const size_t total = (size_t)BATCH * SEQ * KDIM;
  if (i >= total) return;
  int n = (int)(i % KDIM);
  size_t bt = i / KDIM;
  int t = (int)(bt % SEQ), b = (int)(bt / SEQ);
  int r = rep[b * SEQ + t];
  Y[((size_t)b * KDIM + n) * SEQ + t] = (_Float16)X[((size_t)b * SEQ + r) * KDIM + n];
}

// ---------------------------------------------------------------------------
// Row softmax in place over SxS slabs (grid.y = slab)
// ---------------------------------------------------------------------------
__global__ __launch_bounds__(256) void softmax_rows(float* __restrict__ S) {
  float* row = S + ((size_t)blockIdx.y * SEQ + blockIdx.x) * SEQ;
  __shared__ float red[256];
  float m = -INFINITY;
  for (int c = threadIdx.x; c < SEQ; c += 256) m = fmaxf(m, row[c]);
  red[threadIdx.x] = m; __syncthreads();
  for (int st = 128; st > 0; st >>= 1) {
    if (threadIdx.x < st) red[threadIdx.x] = fmaxf(red[threadIdx.x], red[threadIdx.x + st]);
    __syncthreads();
  }
  const float mx = red[0];
  __syncthreads();
  float s = 0.0f;
  for (int c = threadIdx.x; c < SEQ; c += 256) {
    float e = __expf(row[c] - mx); row[c] = e; s += e;
  }
  red[threadIdx.x] = s; __syncthreads();
  for (int st = 128; st > 0; st >>= 1) {
    if (threadIdx.x < st) red[threadIdx.x] += red[threadIdx.x + st];
    __syncthreads();
  }
  const float inv = 1.0f / red[0];
  for (int c = threadIdx.x; c < SEQ; c += 256) row[c] *= inv;
}

// Pbar[b] (+)= (1/H) * sum_{p=0..7} P[p]   -- deterministic (no atomics)
__global__ void pbar_accum(const float* __restrict__ S, float* __restrict__ Pbar,
                           int b, int first) {
  size_t i = (size_t)blockIdx.x * blockDim.x + threadIdx.x;
  const size_t total = (size_t)SEQ * SEQ;
  size_t stride = (size_t)gridDim.x * blockDim.x;
  float* dst = Pbar + (size_t)b * SEQ * SEQ;
  for (; i < total; i += stride) {
    float s = 0.0f;
#pragma unroll
    for (int p = 0; p < 8; ++p) s += S[(size_t)p * SEQ * SEQ + i];
    s *= (1.0f / HEADS);
    if (first) dst[i] = s; else dst[i] += s;
  }
}

// ---------------------------------------------------------------------------
// Host-side orchestration
// ---------------------------------------------------------------------------
extern "C" void kernel_launch(void* const* d_in, const int* in_sizes, int n_in,
                              void* d_out, int out_size, void* d_ws, size_t ws_size,
                              hipStream_t stream) {
  const float* hs  = (const float*)d_in[0];
  const float* Wq  = (const float*)d_in[1];
  const float* bq  = (const float*)d_in[2];
  const float* Wk  = (const float*)d_in[3];
  const float* bk  = (const float*)d_in[4];
  const float* Wv  = (const float*)d_in[5];
  const float* bv  = (const float*)d_in[6];
  const float* Wqk = (const float*)d_in[7];
  const float* bqk = (const float*)d_in[8];
  const float* Wo  = (const float*)d_in[9];
  const float* bo  = (const float*)d_in[10];
  float* out = (float*)d_out;

  char* base = (char*)d_ws;
  size_t off = 0;
  auto carve = [&](size_t bytes) -> void* {
    off = (off + 255) & ~(size_t)255;
    void* p = base + off;
    off += bytes;
    return p;
  };
  const size_t MS = (size_t)BATCH * SEQ;          // 2048 rows
  _Float16* h16    = (_Float16*)carve(MS * DMODEL * 2);
  _Float16* WqT    = (_Float16*)carve((size_t)DMODEL * DMODEL * 2);
  _Float16* WkT    = (_Float16*)carve((size_t)KDIM * DMODEL * 2);
  _Float16* WvT    = (_Float16*)carve((size_t)KDIM * DMODEL * 2);
  _Float16* WqkT   = (_Float16*)carve((size_t)KDIM * QHD * 2);
  _Float16* WoT    = (_Float16*)carve((size_t)DMODEL * KDIM * 2);
  _Float16* q16    = (_Float16*)carve(MS * DMODEL * 2);
  float*    k32    = (float*)   carve(MS * KDIM * 4);
  float*    v32    = (float*)   carve(MS * KDIM * 4);
  _Float16* newk16 = (_Float16*)carve(MS * KDIM * 2);
  _Float16* newvT  = (_Float16*)carve(MS * KDIM * 2);
  int*      repBuf = (int*)     carve(4 * SEQ * 4);
  float*    Pbar   = (float*)   carve((size_t)BATCH * SEQ * SEQ * 4);
  _Float16* pbar16 = (_Float16*)carve((size_t)BATCH * SEQ * SEQ * 2);
  _Float16* ao16   = (_Float16*)carve(MS * KDIM * 2);
  float*    scoreB = (float*)   carve((size_t)8 * SEQ * SEQ * 4);   // also aliases sim
  _Float16* qnBuf  = (_Float16*)carve((size_t)8 * SEQ * KDIM * 2);  // also aliases kn/vn
  // aliases (phase-disjoint):
  float*    simB   = scoreB;                        // 4 slabs of SxS (16.8MB <= 33.5MB)
  _Float16* kn16   = qnBuf;                         // BATCH*SEQ x KDIM
  _Float16* vn16   = qnBuf + MS * KDIM;

  const int TPB = 256;
  auto blocksFor = [](size_t n, int tpb) { return (unsigned)((n + tpb - 1) / tpb); };

  // --- Phase A: precision conversion + weight transposition -----------------
  cvt_f32_to_f16<<<4096, TPB, 0, stream>>>(hs, h16, MS * DMODEL);
  cvt_transpose_f16<<<4096, TPB, 0, stream>>>(Wq,  WqT,  DMODEL, DMODEL);
  cvt_transpose_f16<<<4096, TPB, 0, stream>>>(Wk,  WkT,  DMODEL, KDIM);
  cvt_transpose_f16<<<4096, TPB, 0, stream>>>(Wv,  WvT,  DMODEL, KDIM);
  cvt_transpose_f16<<<1024, TPB, 0, stream>>>(Wqk, WqkT, QHD,    KDIM);
  cvt_transpose_f16<<<4096, TPB, 0, stream>>>(Wo,  WoT,  KDIM,   DMODEL);

  // --- Phase B: QKV projections (WMMA + TDM-staged B) ------------------------
  gemm_wmma_tdm<<<dim3(DMODEL / 64, MS / 128, 1), 128, 0, stream>>>(
      h16, 0, DMODEL, WqT, 0, DMODEL, bq, 1.0f,
      nullptr, 0, q16, 0, DMODEL, DMODEL);
  gemm_wmma_tdm<<<dim3(KDIM / 64, MS / 128, 1), 128, 0, stream>>>(
      h16, 0, DMODEL, WkT, 0, DMODEL, bk, 1.0f,
      k32, 0, nullptr, 0, KDIM, DMODEL);
  gemm_wmma_tdm<<<dim3(KDIM / 64, MS / 128, 1), 128, 0, stream>>>(
      h16, 0, DMODEL, WvT, 0, DMODEL, bv, 1.0f,
      v32, 0, nullptr, 0, KDIM, DMODEL);

  // --- Phase C: dynamic token replacement -----------------------------------
  normalize_rows<<<(unsigned)MS, 256, 0, stream>>>(k32, kn16);
  normalize_rows<<<(unsigned)MS, 256, 0, stream>>>(v32, vn16);
  // simK[b] = kn[b] @ kn[b]^T  (slabs 0,1); simV[b] (slabs 2,3)
  gemm_wmma_tdm<<<dim3(SEQ / 64, SEQ / 128, 2), 128, 0, stream>>>(
      kn16, (size_t)SEQ * KDIM, KDIM, kn16, (size_t)SEQ * KDIM, KDIM, nullptr, 1.0f,
      simB, (size_t)SEQ * SEQ, nullptr, 0, SEQ, KDIM);
  gemm_wmma_tdm<<<dim3(SEQ / 64, SEQ / 128, 2), 128, 0, stream>>>(
      vn16, (size_t)SEQ * KDIM, KDIM, vn16, (size_t)SEQ * KDIM, KDIM, nullptr, 1.0f,
      simB + (size_t)2 * SEQ * SEQ, (size_t)SEQ * SEQ, nullptr, 0, SEQ, KDIM);
  greedy_merge_scan<<<4, SEQ, 0, stream>>>(simB, repBuf);
  gather_rows_f16<<<blocksFor(MS * KDIM, TPB), TPB, 0, stream>>>(k32, repBuf, newk16);
  gather_rows_T_f16<<<blocksFor(MS * KDIM, TPB), TPB, 0, stream>>>(v32, repBuf + 2 * SEQ, newvT);

  // --- Phase D: attention in 8 groups of 8 (b,h) pairs -----------------------
  const float scale = 1.0f / sqrtf((float)KDIM);   // 1/32
  for (int g = 0; g < 8; ++g) {
    const int pair0 = g * 8;
    const int b  = pair0 >> 5;                      // groups 0-3 -> b=0, 4-7 -> b=1
    const int h0 = pair0 & 31;
    // q_new[p] = q[b,:,h0+p,:] @ Wqk + bqk   (M=S, K=96, N=KDIM), f16 out
    gemm_wmma_tdm<<<dim3(KDIM / 64, SEQ / 128, 8), 128, 0, stream>>>(
        q16 + ((size_t)b * SEQ * HEADS + h0) * QHD, (size_t)QHD, HEADS * QHD,
        WqkT, 0, QHD, bqk, 1.0f,
        nullptr, 0, qnBuf, (size_t)SEQ * KDIM, KDIM, QHD);
    // scores[p] = q_new[p] @ new_k[b]^T * scale  (f32 out)
    gemm_wmma_tdm<<<dim3(SEQ / 64, SEQ / 128, 8), 128, 0, stream>>>(
        qnBuf, (size_t)SEQ * KDIM, KDIM,
        newk16 + (size_t)b * SEQ * KDIM, 0, KDIM, nullptr, scale,
        scoreB, (size_t)SEQ * SEQ, nullptr, 0, SEQ, KDIM);
    softmax_rows<<<dim3(SEQ, 8), 256, 0, stream>>>(scoreB);
    pbar_accum<<<2048, TPB, 0, stream>>>(scoreB, Pbar, b, (g % 4 == 0) ? 1 : 0);
  }

  // --- Phase E: out = Pbar @ new_v, then @ Wo + bo ---------------------------
  cvt_f32_to_f16<<<4096, TPB, 0, stream>>>(Pbar, pbar16, (size_t)BATCH * SEQ * SEQ);
  gemm_wmma_tdm<<<dim3(KDIM / 64, SEQ / 128, 2), 128, 0, stream>>>(
      pbar16, (size_t)SEQ * SEQ, SEQ, newvT, (size_t)KDIM * SEQ, SEQ, nullptr, 1.0f,
      nullptr, 0, ao16, (size_t)SEQ * KDIM, KDIM, SEQ);
  gemm_wmma_tdm<<<dim3(DMODEL / 64, MS / 128, 1), 128, 0, stream>>>(
      ao16, 0, KDIM, WoT, 0, KDIM, bo, 1.0f,
      out, 0, nullptr, 0, DMODEL, KDIM);

  (void)in_sizes; (void)n_in; (void)out_size; (void)ws_size;
}
